// GraphHead_29703993819509
// MI455X (gfx1250) — matrix operand order repaired and checked by reading, not compiled, not run on hardware
//
#include <hip/hip_runtime.h>
#include <stdint.h>

// ---------------------------------------------------------------------------
// Types
// ---------------------------------------------------------------------------
typedef __bf16 bf16;
typedef __attribute__((ext_vector_type(16))) __bf16 v16bf;
typedef __attribute__((ext_vector_type(8)))  float  v8f;
typedef __attribute__((ext_vector_type(4)))  int    v4i_t;

union Frag { uint32_t u[8]; v16bf v; };

#define TILE_M 128
#define TILE_N 128
#define TILE_K 32
#define LDS_STR 48   // bf16 elements per LDS row (padded, keeps 16B alignment)

// Probe for the CDNA5 async global->LDS copy path (ASYNCcnt-tracked).
#if defined(__has_builtin)
#if __has_builtin(__builtin_amdgcn_global_load_async_to_lds_b128) && \
    __has_builtin(__builtin_amdgcn_s_wait_asynccnt)
#define USE_ASYNC_LDS 1
#endif
#endif
#ifndef USE_ASYNC_LDS
#define USE_ASYNC_LDS 0
#endif

#if USE_ASYNC_LDS
#define AS1 __attribute__((address_space(1)))
#define AS3 __attribute__((address_space(3)))
// Builtin signature (from toolchain diagnostic): param0 = int4 addrspace(1)*,
// param1 = int4 addrspace(3)*, then imm offset + imm cpol.
__device__ __forceinline__ void async_b128(const bf16* g, bf16* l) {
    __builtin_amdgcn_global_load_async_to_lds_b128(
        (AS1 v4i_t*)g, (AS3 v4i_t*)l, 0, 0);
}
#endif

// Shared inner tile: per-lane fragment gathers (documented CDNA5 VGPR layouts)
// + 2x4 WMMA rank-32 updates.
__device__ __forceinline__ void wmma_tile(const bf16* __restrict__ sAbuf,
                                          const bf16* __restrict__ sBbuf,
                                          int wm, int wn, int lane,
                                          v8f acc[2][4]) {
    Frag fa[2], fb[4];
    // A: lanes 0-15 hold K {0..7,16..23}, lanes 16-31 hold K {8..15,24..31}
    const int arow0 = wm + (lane & 15);
    const int akb   = (lane >> 4) * 8;
#pragma unroll
    for (int i = 0; i < 2; ++i) {
        const bf16* p = &sAbuf[(arow0 + i * 16) * LDS_STR];
#pragma unroll
        for (int d = 0; d < 4; ++d) {
            fa[i].u[d]     = *(const uint32_t*)&p[akb + 2 * d];
            fa[i].u[d + 4] = *(const uint32_t*)&p[16 + akb + 2 * d];
        }
    }
    // B: N = lane&15, lanes 0-15 K {0..7,16..23}, lanes 16-31 K {8..15,24..31}
    const int bkb = (lane >> 4) * 8;
#pragma unroll
    for (int j = 0; j < 4; ++j) {
        const bf16* p = &sBbuf[(wn + j * 16 + (lane & 15)) * LDS_STR + bkb];
#pragma unroll
        for (int d = 0; d < 8; ++d) fb[j].u[d] = *(const uint32_t*)&p[2 * d + (d >= 4 ? 8 : 0)];
    }
#pragma unroll
    for (int i = 0; i < 2; ++i)
#pragma unroll
        for (int j = 0; j < 4; ++j)
            acc[i][j] = __builtin_amdgcn_wmma_f32_16x16x32_bf16(
                false, fa[i].v, false, fb[j].v, (short)0, acc[i][j], false, false);
}

// ---------------------------------------------------------------------------
// Generic bf16 WMMA GEMM:  C = act(A[M,K] @ Bt[N,K]^T + bias[N])
//   Bt is B transposed (row-major [N,K], leading dim ldb).
//   Optional f32 output (ldc, column offset coff, relu) and bf16 output.
//   Requires: M%128==0, N%128==0, K%32==0, 16B-aligned A/Bt rows.
// ---------------------------------------------------------------------------
__global__ __launch_bounds__(256) void gemm_bf16_wmma(
    const bf16* __restrict__ A, int lda,
    const bf16* __restrict__ Bt, int ldb,
    const float* __restrict__ bias,
    float* __restrict__ Cf, int ldc, int coff, int reluF,
    bf16* __restrict__ Cb, int ldcb, int reluB,
    int M, int N, int K)
{
    const int tid  = threadIdx.x;
    const int lane = tid & 31;
    const int wave = tid >> 5;
    const int m0   = blockIdx.y * TILE_M;
    const int n0   = blockIdx.x * TILE_N;
    const int wm   = (wave >> 1) * 32;   // wave row offset within tile
    const int wn   = (wave & 1) * 64;    // wave col offset within tile

    // cooperative tile loaders: 2 threads per row, 32B (16 bf16) each
    const int lrow  = tid >> 1;
    const int lhalf = (tid & 1) * 16;

    v8f acc[2][4] = {};

#if USE_ASYNC_LDS
    // Double-buffered async staging: tile t+1 copies overlap tile t WMMAs.
    __shared__ bf16 sA[2][TILE_M * LDS_STR];
    __shared__ bf16 sB[2][TILE_N * LDS_STR];
    const bf16* gA = A  + (size_t)(m0 + lrow) * lda + lhalf;
    const bf16* gB = Bt + (size_t)(n0 + lrow) * ldb + lhalf;

    auto stage = [&](int buf, int kk) {
        const bf16* ga = gA + kk;
        const bf16* gb = gB + kk;
        bf16* la = &sA[buf][lrow * LDS_STR + lhalf];
        bf16* lb = &sB[buf][lrow * LDS_STR + lhalf];
        async_b128(ga,     la);       // 4 async b128 per thread per stage
        async_b128(ga + 8, la + 8);
        async_b128(gb,     lb);
        async_b128(gb + 8, lb + 8);
    };

    const int nt = K / TILE_K;
    stage(0, 0);
    for (int t = 0; t < nt; ++t) {
        const int buf = t & 1;
        if (t + 1 < nt) {
            stage(buf ^ 1, (t + 1) * TILE_K);
            __builtin_amdgcn_s_wait_asynccnt(4);  // tile t's 4 copies done
        } else {
            __builtin_amdgcn_s_wait_asynccnt(0);
        }
        __syncthreads();                           // all waves' copies visible
        wmma_tile(sA[buf], sB[buf], wm, wn, lane, acc);
        __syncthreads();                           // done reading before reuse
    }
#else
    // Synchronous staging fallback (VGPR round-trip) + prefetch.
    __shared__ bf16 sA[TILE_M * LDS_STR];
    __shared__ bf16 sB[TILE_N * LDS_STR];
    for (int kk = 0; kk < K; kk += TILE_K) {
        const uint4* ga = (const uint4*)(A  + (size_t)(m0 + lrow) * lda + kk + lhalf);
        const uint4* gb = (const uint4*)(Bt + (size_t)(n0 + lrow) * ldb + kk + lhalf);
        uint4 va0 = ga[0], va1 = ga[1];
        uint4 vb0 = gb[0], vb1 = gb[1];
        if (kk + TILE_K < K) {
            __builtin_prefetch(A  + (size_t)(m0 + lrow) * lda + kk + TILE_K + lhalf, 0, 1);
            __builtin_prefetch(Bt + (size_t)(n0 + lrow) * ldb + kk + TILE_K + lhalf, 0, 1);
        }
        __syncthreads();
        { uint4* d = (uint4*)&sA[lrow * LDS_STR + lhalf]; d[0] = va0; d[1] = va1; }
        { uint4* d = (uint4*)&sB[lrow * LDS_STR + lhalf]; d[0] = vb0; d[1] = vb1; }
        __syncthreads();
        wmma_tile(sA, sB, wm, wn, lane, acc);
    }
#endif

    // epilogue: C layout -> lanes 0-15 M=r, lanes 16-31 M=8+r; N = lane&15
    const int rlo = lane >> 4;
    const int cn  = lane & 15;
#pragma unroll
    for (int i = 0; i < 2; ++i) {
#pragma unroll
        for (int j = 0; j < 4; ++j) {
#pragma unroll
            for (int r = 0; r < 8; ++r) {
                const int gm = m0 + wm + i * 16 + r + 8 * rlo;
                const int gn = n0 + wn + j * 16 + cn;
                float v = acc[i][j][r];
                if (bias) v += bias[gn];
                if (Cf) Cf[(size_t)gm * ldc + coff + gn] = reluF ? fmaxf(v, 0.f) : v;
                if (Cb) Cb[(size_t)gm * ldcb + gn] = (bf16)(reluB ? fmaxf(v, 0.f) : v);
            }
        }
    }
}

// ---------------------------------------------------------------------------
// Data-prep / elementwise kernels
// ---------------------------------------------------------------------------
__global__ void k_f32_to_bf16(const float* __restrict__ in, bf16* __restrict__ out, int n) {
    int i = blockIdx.x * blockDim.x + threadIdx.x;
    if (i < n) out[i] = (bf16)in[i];
}

// out[N,Kpad] (bf16) = transpose(in[Kin,N] f32), zero-padded in K
__global__ void k_transpose_pad(const float* __restrict__ in, bf16* __restrict__ out,
                                int Kin, int Kpad, int N) {
    int i = blockIdx.x * blockDim.x + threadIdx.x;
    if (i >= N * Kpad) return;
    int n = i / Kpad, k = i - n * Kpad;
    out[i] = (k < Kin) ? (bf16)in[(size_t)k * N + n] : (bf16)0.f;
}

// in [16, din, 64] f32 -> out [(c*64+t), din] bf16
__global__ void k_branch_w_t(const float* __restrict__ in, bf16* __restrict__ out, int din) {
    int i = blockIdx.x * blockDim.x + threadIdx.x;
    if (i >= 1024 * din) return;
    int row = i / din, k = i - row * din;
    int c = row >> 6, t = row & 63;
    out[i] = (bf16)in[((size_t)c * din + k) * 64 + t];
}

// pair_spatial36 [P,36] -> bf16 [P,64] zero-padded
__global__ void k_sp_pad(const float* __restrict__ in, bf16* __restrict__ out) {
    int i = blockIdx.x * blockDim.x + threadIdx.x;   // P*64
    int p = i >> 6, k = i & 63;
    out[i] = (k < 36) ? (bf16)in[p * 36 + k] : (bf16)0.f;
}

__global__ void k_b3sum(const float* __restrict__ b3, float* __restrict__ out) {
    int j = blockIdx.x * blockDim.x + threadIdx.x;
    if (j >= 1024) return;
    float s = 0.f;
    for (int c = 0; c < 16; ++c) s += b3[c * 1024 + j];
    out[j] = s;
}

__global__ void k_pool_g(const float* __restrict__ feat3, float* __restrict__ g) {
    int ch = threadIdx.x;                 // 256 threads
    const float* p = feat3 + ch * 625;
    float s = 0.f;
    for (int i = 0; i < 625; ++i) s += p[i];
    g[ch] = s * (1.f / 625.f);
}

// a_g[j=(c,t)] = g[256] . attg.w1[c,:,t] + attg.b1[c,t]
__global__ void k_ag(const float* __restrict__ g, const float* __restrict__ w1,
                     const float* __restrict__ b1, float* __restrict__ ag) {
    int j = blockIdx.x * blockDim.x + threadIdx.x;
    if (j >= 1024) return;
    int c = j >> 6, t = j & 63;
    const float* w = w1 + (size_t)c * 256 * 64 + t;
    float s = b1[j];
    for (int k = 0; k < 256; ++k) s += g[k] * w[k * 64];
    ag[j] = s;
}

// h2 = relu((U[x(p)] + V[y(p)] + b1) * L) -> bf16   (P*1024 elems)
__global__ void k_h2_att(const float* __restrict__ U, const float* __restrict__ V,
                         const float* __restrict__ b1, const bf16* __restrict__ L,
                         bf16* __restrict__ h2) {
    int i = blockIdx.x * blockDim.x + threadIdx.x;
    int p = i >> 10, j = i & 1023;
    int xi = p >> 7, yi = p & 127;
    float v = (U[xi * 1024 + j] + V[yi * 1024 + j] + b1[j]) * (float)L[i];
    h2[i] = (bf16)fmaxf(v, 0.f);
}

__global__ void k_h2_attg(const float* __restrict__ ag, const bf16* __restrict__ L,
                          bf16* __restrict__ h2) {
    int i = blockIdx.x * blockDim.x + threadIdx.x;
    int j = i & 1023;
    float v = ag[j] * (float)L[i];
    h2[i] = (bf16)fmaxf(v, 0.f);
}

// adj[row] = dot(watt_bf[row,:1024], adj_w) + adj_b   (one wave32 per row)
__global__ __launch_bounds__(256) void k_adj(const bf16* __restrict__ w,
                                             const float* __restrict__ adjw,
                                             const float* __restrict__ adjb,
                                             float* __restrict__ adj) {
    int row = blockIdx.x * 8 + (threadIdx.x >> 5);
    int lane = threadIdx.x & 31;
    const bf16* r = w + (size_t)row * 1024;
    float s = 0.f;
    for (int j = lane; j < 1024; j += 32) s += (float)r[j] * adjw[j];
    for (int o = 16; o > 0; o >>= 1) s += __shfl_down(s, o, 32);
    if (lane == 0) adj[row] = s + adjb[0];
}

__global__ void k_softmax_rows(const float* __restrict__ adj, float* __restrict__ out) {
    __shared__ float red[128];
    int s = blockIdx.x, t = threadIdx.x;
    float x = adj[s * 128 + t];
    red[t] = x; __syncthreads();
    for (int o = 64; o > 0; o >>= 1) { if (t < o) red[t] = fmaxf(red[t], red[t + o]); __syncthreads(); }
    float mx = red[0]; __syncthreads();
    float e = __expf(x - mx);
    red[t] = e; __syncthreads();
    for (int o = 64; o > 0; o >>= 1) { if (t < o) red[t] += red[t + o]; __syncthreads(); }
    out[s * 128 + t] = e / red[0];
}

__global__ void k_softmax_cols(const float* __restrict__ adj, float* __restrict__ out) {
    __shared__ float red[128];
    int o = blockIdx.x, t = threadIdx.x;        // t indexes subject s
    float x = adj[t * 128 + o];
    red[t] = x; __syncthreads();
    for (int q = 64; q > 0; q >>= 1) { if (t < q) red[t] = fmaxf(red[t], red[t + q]); __syncthreads(); }
    float mx = red[0]; __syncthreads();
    float e = __expf(x - mx);
    red[t] = e; __syncthreads();
    for (int q = 64; q > 0; q >>= 1) { if (t < q) red[t] += red[t + q]; __syncthreads(); }
    out[o * 128 + t] = e / red[0];
}

// T_s[s,:] = sum_o Arow[s,o] * relu(A1[o,:] * L[(s*128+o),:])   -> bf16
__global__ __launch_bounds__(256) void k_wsum_o2s(const float* __restrict__ Arow,
                                                  const float* __restrict__ A1,
                                                  const bf16* __restrict__ L,
                                                  bf16* __restrict__ T) {
    int s = blockIdx.x, t = threadIdx.x;
    float acc[4] = {0.f, 0.f, 0.f, 0.f};
    for (int o = 0; o < 128; ++o) {
        float w = Arow[s * 128 + o];
        const float* a = A1 + o * 1024;
        const bf16* l = L + (size_t)(s * 128 + o) * 1024;
#pragma unroll
        for (int q = 0; q < 4; ++q) {
            int j = t + q * 256;
            acc[q] += w * fmaxf(a[j] * (float)l[j], 0.f);
        }
    }
#pragma unroll
    for (int q = 0; q < 4; ++q) T[s * 1024 + t + q * 256] = (bf16)acc[q];
}

// T_o[o,:] = sum_s Bcol[o,s] * relu(A1[s,:] * L[(s*128+o),:])   -> bf16
__global__ __launch_bounds__(256) void k_wsum_s2o(const float* __restrict__ Bcol,
                                                  const float* __restrict__ A1,
                                                  const bf16* __restrict__ L,
                                                  bf16* __restrict__ T) {
    int o = blockIdx.x, t = threadIdx.x;
    float acc[4] = {0.f, 0.f, 0.f, 0.f};
    for (int s = 0; s < 128; ++s) {
        float w = Bcol[o * 128 + s];
        const float* a = A1 + s * 1024;
        const bf16* l = L + (size_t)(s * 128 + o) * 1024;
#pragma unroll
        for (int q = 0; q < 4; ++q) {
            int j = t + q * 256;
            acc[q] += w * fmaxf(a[j] * (float)l[j], 0.f);
        }
    }
#pragma unroll
    for (int q = 0; q < 4; ++q) T[o * 1024 + t + q * 256] = (bf16)acc[q];
}

// h = LN(h + m) * g + b ; also write bf16 mirror
__global__ __launch_bounds__(256) void k_ln(float* __restrict__ h, bf16* __restrict__ hb,
                                            const float* __restrict__ m,
                                            const float* __restrict__ gam,
                                            const float* __restrict__ bet) {
    __shared__ float red[256];
    int row = blockIdx.x, t = threadIdx.x;
    float x[4]; float s = 0.f;
#pragma unroll
    for (int q = 0; q < 4; ++q) {
        int j = t + q * 256;
        x[q] = h[row * 1024 + j] + m[row * 1024 + j];
        s += x[q];
    }
    red[t] = s; __syncthreads();
    for (int o = 128; o > 0; o >>= 1) { if (t < o) red[t] += red[t + o]; __syncthreads(); }
    float mean = red[0] * (1.f / 1024.f); __syncthreads();
    float v = 0.f;
#pragma unroll
    for (int q = 0; q < 4; ++q) { float d = x[q] - mean; v += d * d; }
    red[t] = v; __syncthreads();
    for (int o = 128; o > 0; o >>= 1) { if (t < o) red[t] += red[t + o]; __syncthreads(); }
    float inv = rsqrtf(red[0] * (1.f / 1024.f) + 1e-5f);
#pragma unroll
    for (int q = 0; q < 4; ++q) {
        int j = t + q * 256;
        float y = (x[q] - mean) * inv * gam[j] + bet[j];
        h[row * 1024 + j] = y;
        hb[row * 1024 + j] = (bf16)y;
    }
}

// ---------------------------------------------------------------------------
// Host-side orchestration
// ---------------------------------------------------------------------------
extern "C" void kernel_launch(void* const* d_in, const int* in_sizes, int n_in,
                              void* d_out, int out_size, void* d_ws, size_t ws_size,
                              hipStream_t stream) {
    (void)in_sizes; (void)n_in; (void)out_size; (void)ws_size;
    const int P = 16384;

    const float* feat3  = (const float*)d_in[0];
    const float* subj   = (const float*)d_in[1];
    const float* obj    = (const float*)d_in[2];
    const float* sp36   = (const float*)d_in[3];
    const float* bh_w1  = (const float*)d_in[4];
    const float* bh_b1  = (const float*)d_in[5];
    const float* bh_w2  = (const float*)d_in[6];
    const float* bh_b2  = (const float*)d_in[7];
    const float* sp_w1  = (const float*)d_in[8];
    const float* sp_b1  = (const float*)d_in[9];
    const float* sp_w2  = (const float*)d_in[10];
    const float* sp_b2  = (const float*)d_in[11];
    const float* sp_w3  = (const float*)d_in[12];
    const float* sp_b3  = (const float*)d_in[13];
    const float* adj_w  = (const float*)d_in[14];
    const float* adj_b  = (const float*)d_in[15];
    const float* ln_s_g = (const float*)d_in[16];
    const float* ln_s_b = (const float*)d_in[17];
    const float* ln_o_g = (const float*)d_in[18];
    const float* ln_o_b = (const float*)d_in[19];
    // mbf param sets: w1,b1,w2,b2,w3,b3
    const float* o2s_w1 = (const float*)d_in[20]; const float* o2s_b1 = (const float*)d_in[21];
    const float* o2s_w2 = (const float*)d_in[22]; const float* o2s_b2 = (const float*)d_in[23];
    const float* o2s_w3 = (const float*)d_in[24]; const float* o2s_b3 = (const float*)d_in[25];
    const float* s2o_w1 = (const float*)d_in[26]; const float* s2o_b1 = (const float*)d_in[27];
    const float* s2o_w2 = (const float*)d_in[28]; const float* s2o_b2 = (const float*)d_in[29];
    const float* s2o_w3 = (const float*)d_in[30]; const float* s2o_b3 = (const float*)d_in[31];
    const float* att_w1 = (const float*)d_in[32]; const float* att_b1 = (const float*)d_in[33];
    const float* att_w2 = (const float*)d_in[34]; const float* att_b2 = (const float*)d_in[35];
    const float* att_w3 = (const float*)d_in[36]; const float* att_b3 = (const float*)d_in[37];
    const float* atg_w1 = (const float*)d_in[38]; const float* atg_b1 = (const float*)d_in[39];
    const float* atg_w2 = (const float*)d_in[40]; const float* atg_b2 = (const float*)d_in[41];
    const float* atg_w3 = (const float*)d_in[42]; const float* atg_b3 = (const float*)d_in[43];
    float* out = (float*)d_out;

    // ---- workspace bump allocator (256B aligned) ----
    char* wsb = (char*)d_ws;
    size_t off = 0;
    auto alloc = [&](size_t bytes) -> void* {
        off = (off + 255) & ~(size_t)255;
        void* p = wsb + off;
        off += bytes;
        return p;
    };
    bf16*  soj_bf  = (bf16*)alloc((size_t)256 * 12544 * 2);
    bf16*  bhw1t   = (bf16*)alloc((size_t)1024 * 12544 * 2);
    bf16*  bhw2t   = (bf16*)alloc((size_t)1024 * 1024 * 2);
    bf16*  hb1     = (bf16*)alloc((size_t)256 * 1024 * 2);
    float* hso     = (float*)alloc((size_t)256 * 1024 * 4);
    bf16*  hso_bf  = (bf16*)alloc((size_t)256 * 1024 * 2);
    bf16*  sp64    = (bf16*)alloc((size_t)P * 64 * 2);
    bf16*  spw1t   = (bf16*)alloc((size_t)128 * 64 * 2);
    bf16*  spw2t   = (bf16*)alloc((size_t)256 * 128 * 2);
    bf16*  spw3t   = (bf16*)alloc((size_t)1024 * 256 * 2);
    bf16*  sp1     = (bf16*)alloc((size_t)P * 128 * 2);
    bf16*  sp2     = (bf16*)alloc((size_t)P * 256 * 2);
    bf16*  spb     = (bf16*)alloc((size_t)P * 1024 * 2);
    bf16*  W2attT  = (bf16*)alloc((size_t)1024 * 1024 * 2);
    bf16*  W2o2sT  = (bf16*)alloc((size_t)1024 * 1024 * 2);
    bf16*  W2s2oT  = (bf16*)alloc((size_t)1024 * 1024 * 2);
    bf16*  W2atgT  = (bf16*)alloc((size_t)1024 * 1024 * 2);
    bf16*  W1attT  = (bf16*)alloc((size_t)1024 * 2048 * 2);
    bf16*  W1o2sT  = (bf16*)alloc((size_t)1024 * 1024 * 2);
    bf16*  W1s2oT  = (bf16*)alloc((size_t)1024 * 1024 * 2);
    bf16*  W3attT  = (bf16*)alloc((size_t)1024 * 1024 * 2);
    bf16*  W3o2sT  = (bf16*)alloc((size_t)1024 * 1024 * 2);
    bf16*  W3s2oT  = (bf16*)alloc((size_t)1024 * 1024 * 2);
    bf16*  W3atgT  = (bf16*)alloc((size_t)1024 * 1024 * 2);
    float* b3s_att = (float*)alloc(1024 * 4);
    float* b3s_o2s = (float*)alloc(1024 * 4);
    float* b3s_s2o = (float*)alloc(1024 * 4);
    float* b3s_atg = (float*)alloc(1024 * 4);
    bf16*  L_att   = (bf16*)alloc((size_t)P * 1024 * 2);
    bf16*  L_o2s   = (bf16*)alloc((size_t)P * 1024 * 2);
    bf16*  L_s2o   = (bf16*)alloc((size_t)P * 1024 * 2);
    bf16*  L_atg   = (bf16*)alloc((size_t)P * 1024 * 2);
    float* U       = (float*)alloc((size_t)128 * 1024 * 4);
    float* V       = (float*)alloc((size_t)128 * 1024 * 4);
    float* A1      = (float*)alloc((size_t)128 * 1024 * 4);
    float* ag      = (float*)alloc(1024 * 4);
    float* gvec    = (float*)alloc(256 * 4);
    bf16*  h2      = (bf16*)alloc((size_t)P * 1024 * 2);
    bf16*  watt_bf = (bf16*)alloc((size_t)P * 1024 * 2);
    float* adj     = (float*)alloc((size_t)P * 4);
    float* Arow    = (float*)alloc((size_t)P * 4);
    float* Bcol    = (float*)alloc((size_t)P * 4);
    bf16*  T       = (bf16*)alloc((size_t)128 * 1024 * 2);
    float* mbuf    = (float*)alloc((size_t)128 * 1024 * 4);

    float* hs = hso;                 bf16* hs_bf = hso_bf;
    float* ho = hso + 128 * 1024;    bf16* ho_bf = hso_bf + 128 * 1024;

    auto gemm = [&](const bf16* A, int lda, const bf16* Bt, int ldb, const float* bias,
                    float* Cf, int ldc, int coff, int reluF,
                    bf16* Cb, int ldcb, int reluB, int M, int N, int K) {
        dim3 grid(N / 128, M / 128);
        gemm_bf16_wmma<<<grid, 256, 0, stream>>>(A, lda, Bt, ldb, bias,
                                                 Cf, ldc, coff, reluF,
                                                 Cb, ldcb, reluB, M, N, K);
    };
    auto launch1d = [&](int n) { return dim3((n + 255) / 256); };

    // ---- weight / input preparation (bf16, transposed) ----
    k_f32_to_bf16<<<launch1d(128 * 12544), 256, 0, stream>>>(subj, soj_bf, 128 * 12544);
    k_f32_to_bf16<<<launch1d(128 * 12544), 256, 0, stream>>>(obj, soj_bf + (size_t)128 * 12544, 128 * 12544);
    k_transpose_pad<<<launch1d(1024 * 12544), 256, 0, stream>>>(bh_w1, bhw1t, 12544, 12544, 1024);
    k_transpose_pad<<<launch1d(1024 * 1024), 256, 0, stream>>>(bh_w2, bhw2t, 1024, 1024, 1024);
    k_transpose_pad<<<launch1d(128 * 64), 256, 0, stream>>>(sp_w1, spw1t, 36, 64, 128);
    k_transpose_pad<<<launch1d(256 * 128), 256, 0, stream>>>(sp_w2, spw2t, 128, 128, 256);
    k_transpose_pad<<<launch1d(1024 * 256), 256, 0, stream>>>(sp_w3, spw3t, 256, 256, 1024);
    k_branch_w_t<<<launch1d(1024 * 1024), 256, 0, stream>>>(att_w2, W2attT, 1024);
    k_branch_w_t<<<launch1d(1024 * 1024), 256, 0, stream>>>(o2s_w2, W2o2sT, 1024);
    k_branch_w_t<<<launch1d(1024 * 1024), 256, 0, stream>>>(s2o_w2, W2s2oT, 1024);
    k_branch_w_t<<<launch1d(1024 * 1024), 256, 0, stream>>>(atg_w2, W2atgT, 1024);
    k_branch_w_t<<<launch1d(1024 * 2048), 256, 0, stream>>>(att_w1, W1attT, 2048);
    k_branch_w_t<<<launch1d(1024 * 1024), 256, 0, stream>>>(o2s_w1, W1o2sT, 1024);
    k_branch_w_t<<<launch1d(1024 * 1024), 256, 0, stream>>>(s2o_w1, W1s2oT, 1024);
    // w3 [16,64,1024] is flat [1024(ct),1024(j)] -> transpose to [j, ct]
    k_transpose_pad<<<launch1d(1024 * 1024), 256, 0, stream>>>(att_w3, W3attT, 1024, 1024, 1024);
    k_transpose_pad<<<launch1d(1024 * 1024), 256, 0, stream>>>(o2s_w3, W3o2sT, 1024, 1024, 1024);
    k_transpose_pad<<<launch1d(1024 * 1024), 256, 0, stream>>>(s2o_w3, W3s2oT, 1024, 1024, 1024);
    k_transpose_pad<<<launch1d(1024 * 1024), 256, 0, stream>>>(atg_w3, W3atgT, 1024, 1024, 1024);
    k_b3sum<<<4, 256, 0, stream>>>(att_b3, b3s_att);
    k_b3sum<<<4, 256, 0, stream>>>(o2s_b3, b3s_o2s);
    k_b3sum<<<4, 256, 0, stream>>>(s2o_b3, b3s_s2o);
    k_b3sum<<<4, 256, 0, stream>>>(atg_b3, b3s_atg);
    k_sp_pad<<<launch1d(P * 64), 256, 0, stream>>>(sp36, sp64);
    k_pool_g<<<1, 256, 0, stream>>>(feat3, gvec);

    // ---- box head (subj+obj stacked: 256 rows) ----
    gemm(soj_bf, 12544, bhw1t, 12544, bh_b1, nullptr, 0, 0, 0, hb1, 1024, 1, 256, 1024, 12544);
    gemm(hb1, 1024, bhw2t, 1024, bh_b2, hso, 1024, 0, 1, hso_bf, 1024, 1, 256, 1024, 1024);

    // ---- spatial MLP ----
    gemm(sp64, 64, spw1t, 64, sp_b1, nullptr, 0, 0, 0, sp1, 128, 1, P, 128, 64);
    gemm(sp1, 128, spw2t, 128, sp_b2, nullptr, 0, 0, 0, sp2, 256, 1, P, 256, 128);
    gemm(sp2, 256, spw3t, 256, sp_b3, nullptr, 0, 0, 0, spb, 1024, 1, P, 1024, 256);

    // ---- iteration-invariant lin2(sp) for all four MBF heads (no relu) ----
    gemm(spb, 1024, W2attT, 1024, att_b2, nullptr, 0, 0, 0, L_att, 1024, 0, P, 1024, 1024);
    gemm(spb, 1024, W2o2sT, 1024, o2s_b2, nullptr, 0, 0, 0, L_o2s, 1024, 0, P, 1024, 1024);
    gemm(spb, 1024, W2s2oT, 1024, s2o_b2, nullptr, 0, 0, 0, L_s2o, 1024, 0, P, 1024, 1024);
    gemm(spb, 1024, W2atgT, 1024, atg_b2, nullptr, 0, 0, 0, L_atg, 1024, 0, P, 1024, 1024);

    auto compute_h2_att = [&]() {
        gemm(hs_bf, 1024, W1attT, 2048, nullptr, U, 1024, 0, 0, nullptr, 0, 0, 128, 1024, 1024);
        gemm(ho_bf, 1024, W1attT + 1024, 2048, nullptr, V, 1024, 0, 0, nullptr, 0, 0, 128, 1024, 1024);
        k_h2_att<<<launch1d(P * 1024), 256, 0, stream>>>(U, V, att_b1, L_att, h2);
    };

    // ---- message-passing iterations ----
    for (int it = 0; it < 2; ++it) {
        compute_h2_att();
        gemm(h2, 1024, W3attT, 1024, b3s_att, nullptr, 0, 0, 0, watt_bf, 1024, 1, P, 1024, 1024);
        k_adj<<<P / 8, 256, 0, stream>>>(watt_bf, adj_w, adj_b, adj);
        k_softmax_rows<<<128, 128, 0, stream>>>(adj, Arow);
        k_softmax_cols<<<128, 128, 0, stream>>>(adj, Bcol);

        // subject update: messages from objects
        gemm(ho_bf, 1024, W1o2sT, 1024, o2s_b1, A1, 1024, 0, 0, nullptr, 0, 0, 128, 1024, 1024);
        k_wsum_o2s<<<128, 256, 0, stream>>>(Arow, A1, L_o2s, T);
        gemm(T, 1024, W3o2sT, 1024, b3s_o2s, mbuf, 1024, 0, 1, nullptr, 0, 0, 128, 1024, 1024);
        k_ln<<<128, 256, 0, stream>>>(hs, hs_bf, mbuf, ln_s_g, ln_s_b);

        // object update: messages from (updated) subjects
        gemm(hs_bf, 1024, W1s2oT, 1024, s2o_b1, A1, 1024, 0, 0, nullptr, 0, 0, 128, 1024, 1024);
        k_wsum_s2o<<<128, 256, 0, stream>>>(Bcol, A1, L_s2o, T);
        gemm(T, 1024, W3s2oT, 1024, b3s_s2o, mbuf, 1024, 0, 1, nullptr, 0, 0, 128, 1024, 1024);
        k_ln<<<128, 256, 0, stream>>>(ho, ho_bf, mbuf, ln_o_g, ln_o_b);
    }

    // ---- final pair features: [att | attg] into [P, 2048] f32 ----
    compute_h2_att();
    gemm(h2, 1024, W3attT, 1024, b3s_att, out, 2048, 0, 1, nullptr, 0, 0, P, 1024, 1024);

    k_ag<<<4, 256, 0, stream>>>(gvec, atg_w1, atg_b1, ag);
    k_h2_attg<<<launch1d(P * 1024), 256, 0, stream>>>(ag, L_atg, h2);
    gemm(h2, 1024, W3atgT, 1024, b3s_atg, out, 2048, 1024, 1, nullptr, 0, 0, P, 1024, 1024);
}
